// Attention_28776280883332
// MI455X (gfx1250) — matrix operand order repaired and checked by reading, not compiled, not run on hardware
//
#include <hip/hip_runtime.h>
#include <hip/hip_bf16.h>

// Problem constants (B=8, N=1024, C=768, H=12, D=64)
#define BDIM   8
#define NSEQ   1024
#define CDIM   768
#define HHEADS 12
#define DHEAD  64
#define MROWS  (BDIM * NSEQ)   // 8192
#define QKVLD  (3 * CDIM)      // 2304

typedef __attribute__((ext_vector_type(16))) _Float16 v16h;
typedef __attribute__((ext_vector_type(8)))  float    v8f;
typedef __attribute__((ext_vector_type(4)))  float    f4;
typedef __attribute__((ext_vector_type(4)))  _Float16 h4;
typedef __attribute__((ext_vector_type(4)))  unsigned u32x4;
typedef __attribute__((ext_vector_type(8)))  int      i32x8;
typedef __attribute__((ext_vector_type(4)))  int      i32x4;

union FragA { v16h v; uint4 q[2]; };
union U16h  { uint4 q[2]; _Float16 h[16]; };
union Fi    { float f; int i; };

__device__ __forceinline__ v8f wmma32(v16h a, v16h b, v8f c) {
  // D = A(16x32 f16) * B(32x16 f16) + C(16x16 f32)
  return __builtin_amdgcn_wmma_f32_16x16x32_f16(
      /*neg_a=*/false, a, /*neg_b=*/false, b,
      /*c_mod=*/(short)0, c, /*reuse_a=*/false, /*reuse_b=*/false);
}

// ---- VALU 16-lane butterfly reductions via DPP (no LDS, no waits) ----------
template <int CTRL>
__device__ __forceinline__ float dpp_mov(float x) {
  Fi s, d;
  s.f = x;
  d.i = __builtin_amdgcn_update_dpp(0, s.i, CTRL, 0xf, 0xf, true);
  return d.f;
}
__device__ __forceinline__ float red_max16(float x) {
  x = fmaxf(x, dpp_mov<0xB1>(x));   // quad_perm [1,0,3,2]  (xor 1)
  x = fmaxf(x, dpp_mov<0x4E>(x));   // quad_perm [2,3,0,1]  (xor 2)
  x = fmaxf(x, dpp_mov<0x141>(x));  // row_half_mirror      (combine 4s -> 8)
  x = fmaxf(x, dpp_mov<0x140>(x));  // row_mirror           (combine 8s -> 16)
  return x;
}
__device__ __forceinline__ float red_sum16(float x) {
  x += dpp_mov<0xB1>(x);
  x += dpp_mov<0x4E>(x);
  x += dpp_mov<0x141>(x);
  x += dpp_mov<0x140>(x);
  return x;
}

// ---- Tensor Data Mover: 2D tile load Global -> LDS -------------------------
#if __has_builtin(__builtin_amdgcn_tensor_load_to_lds)
#define HAVE_TDM 1
__device__ __forceinline__ void tdm_load_2d(u32x4 g0, i32x8 g1) {
  i32x4 z4 = {0, 0, 0, 0};
#if defined(__clang_major__) && (__clang_major__ >= 23)
  i32x8 z8 = {0, 0, 0, 0, 0, 0, 0, 0};
  __builtin_amdgcn_tensor_load_to_lds(g0, g1, z4, z4, z8, 0);
#else
  __builtin_amdgcn_tensor_load_to_lds(g0, g1, z4, z4, 0);
#endif
}
#else
#define HAVE_TDM 0
#endif

// ---------------------------------------------------------------------------
// GEMM: Out[M,N] = A[M,K] * W[N,K]^T + bias[N]
//   A_HALF: A is f16 (attn ws) else f32 (x) converted while staging to LDS
//   OUT_F32: write fp32 (final output) else f16 (qkv ws)
// Block = 256 threads (8 waves), tile 128x128, K-step 32.
// ---------------------------------------------------------------------------
template <bool A_HALF, bool OUT_F32>
__global__ void __launch_bounds__(256)
gemm_wmma(const void* __restrict__ Aptr, const float* __restrict__ W,
          const float* __restrict__ bias, void* __restrict__ Out,
          int M, int K, int Nout)
{
  constexpr int KT = 32;
  constexpr int AS = 40;                  // LDS row stride in halfs (16B-aligned pad)
  __shared__ _Float16 sA[128 * AS];
  __shared__ _Float16 sW[128 * AS];

  const int tid  = threadIdx.x;
  const int lane = tid & 31;
  const int wave = tid >> 5;
  const int wm   = wave & 1;              // 2 waves along M (64 rows each)
  const int wn   = wave >> 1;             // 4 waves along N (32 cols each)
  const int n16  = lane & 15;
  const int g    = lane >> 4;
  const int bm   = blockIdx.y * 128;
  const int bn   = blockIdx.x * 128;

  const v8f vzero = {0.f, 0.f, 0.f, 0.f, 0.f, 0.f, 0.f, 0.f};
  v8f acc[4][2];
#pragma unroll
  for (int mf = 0; mf < 4; ++mf)
#pragma unroll
    for (int nf = 0; nf < 2; ++nf) acc[mf][nf] = vzero;

  const int srow = tid >> 1;              // staging row 0..127
  const int scb  = (tid & 1) * 16;        // staging col chunk (16 elems)

  for (int k0 = 0; k0 < K; k0 += KT) {
    // ---- stage A tile (128 x 32) ----
    if (A_HALF) {
      const _Float16* src = (const _Float16*)Aptr + (size_t)(bm + srow) * K + k0 + scb;
      *(uint4*)&sA[srow * AS + scb]     = *(const uint4*)(src);
      *(uint4*)&sA[srow * AS + scb + 8] = *(const uint4*)(src + 8);
    } else {
      const float* src = (const float*)Aptr + (size_t)(bm + srow) * K + k0 + scb;
#pragma unroll
      for (int j = 0; j < 4; ++j) {
        f4 fv = *(const f4*)(src + 4 * j);
        h4 hv;
        hv[0] = (_Float16)fv[0]; hv[1] = (_Float16)fv[1];
        hv[2] = (_Float16)fv[2]; hv[3] = (_Float16)fv[3];
        *(h4*)&sA[srow * AS + scb + 4 * j] = hv;
      }
    }
    // ---- stage W tile (128 output-cols x 32), convert f32 -> f16 ----
    {
      const float* src = W + (size_t)(bn + srow) * K + k0 + scb;
#pragma unroll
      for (int j = 0; j < 4; ++j) {
        f4 fv = *(const f4*)(src + 4 * j);
        h4 hv;
        hv[0] = (_Float16)fv[0]; hv[1] = (_Float16)fv[1];
        hv[2] = (_Float16)fv[2]; hv[3] = (_Float16)fv[3];
        *(h4*)&sW[srow * AS + scb + 4 * j] = hv;
      }
    }
    __syncthreads();

    // ---- fragments: A (row-striped per lane), B^T (lane = out col) ----
    FragA af[4], bf[2];
#pragma unroll
    for (int mf = 0; mf < 4; ++mf) {
      const _Float16* p = &sA[(wm * 64 + mf * 16 + n16) * AS + g * 8];
      af[mf].q[0] = *(const uint4*)(p);
      af[mf].q[1] = *(const uint4*)(p + 16);
    }
#pragma unroll
    for (int nf = 0; nf < 2; ++nf) {
      const _Float16* p = &sW[(wn * 32 + nf * 16 + n16) * AS + g * 8];
      bf[nf].q[0] = *(const uint4*)(p);
      bf[nf].q[1] = *(const uint4*)(p + 16);
    }
#pragma unroll
    for (int mf = 0; mf < 4; ++mf)
#pragma unroll
      for (int nf = 0; nf < 2; ++nf)
        acc[mf][nf] = wmma32(af[mf].v, bf[nf].v, acc[mf][nf]);
    __syncthreads();
  }

  // ---- epilogue: bias + store ----
#pragma unroll
  for (int nf = 0; nf < 2; ++nf) {
    const int col = bn + wn * 32 + nf * 16 + n16;
    const float bv = bias[col];
#pragma unroll
    for (int mf = 0; mf < 4; ++mf) {
#pragma unroll
      for (int rr = 0; rr < 8; ++rr) {
        const int row = bm + wm * 64 + mf * 16 + rr + 8 * g;
        const float val = acc[mf][nf][rr] + bv;
        if constexpr (OUT_F32)
          ((float*)Out)[(size_t)row * Nout + col] = val;
        else
          ((_Float16*)Out)[(size_t)row * Nout + col] = (_Float16)val;
      }
    }
  }
}

// ---------------------------------------------------------------------------
// Flash attention: one block per (b, h, 128-query chunk); 8 waves x 16 rows.
// qkv ws layout: f16 [8192][2304] = row m, cols [Q | K | V] x 12 heads x 64.
// attn ws layout: f16 [8192][768].
// K tile staged by the Tensor Data Mover (TDM) with LDS padding giving the
// 72-half row stride; V tile staged transposed by the block.
// ---------------------------------------------------------------------------
__global__ void __launch_bounds__(256)
flash_attn(const _Float16* __restrict__ qkv, _Float16* __restrict__ attn)
{
  constexpr int LS = 72;                  // LDS row stride (16B-aligned pad)
  __shared__ _Float16 sK [64 * LS];       // [key][d]   (TDM destination)
  __shared__ _Float16 sVT[64 * LS];       // [d][key]   (transposed at staging)
  __shared__ _Float16 sP [8 * 16 * LS];   // per-wave P bounce, [qrow][key]

  const int tid  = threadIdx.x;
  const int lane = tid & 31;
  const int wave = tid >> 5;
  const int n16  = lane & 15;
  const int g    = lane >> 4;

  const int bh    = blockIdx.x;
  const int b     = bh / HHEADS;
  const int h     = bh % HHEADS;
  const int mbase = b * NSEQ;
  const int grow  = mbase + blockIdx.y * 128 + wave * 16;  // first query row (global)
  const size_t qoff = (size_t)h * DHEAD;
  const size_t koff = CDIM + qoff;
  const size_t voff = 2 * (size_t)CDIM + qoff;

  // ---- load Q fragments (16 rows x 64 d) and fold in softmax scale ----
  FragA qf[2];
#pragma unroll
  for (int kk = 0; kk < 2; ++kk) {
    const _Float16* p = qkv + (size_t)(grow + n16) * QKVLD + qoff + kk * 32 + g * 8;
    qf[kk].q[0] = *(const uint4*)(p);
    qf[kk].q[1] = *(const uint4*)(p + 16);
  }
  const _Float16 sc = (_Float16)0.125f;   // 1/sqrt(64)
#pragma unroll
  for (int kk = 0; kk < 2; ++kk)
#pragma unroll
    for (int i = 0; i < 16; ++i) qf[kk].v[i] *= sc;

  const v8f vzero = {0.f, 0.f, 0.f, 0.f, 0.f, 0.f, 0.f, 0.f};
  v8f o[4];
#pragma unroll
  for (int c = 0; c < 4; ++c) o[c] = vzero;
  float mrow[8], lrow[8];
#pragma unroll
  for (int rr = 0; rr < 8; ++rr) { mrow[rr] = -1.0e30f; lrow[rr] = 0.f; }

  _Float16* myP = sP + wave * 16 * LS;
  const int srow = tid >> 2;              // staging key row 0..63
  const int scb  = (tid & 3) * 16;        // staging d chunk (16 halfs)

#if HAVE_TDM
  const unsigned lds_k = (unsigned)(size_t)(void*)&sK[0];
#endif

  for (int kb = 0; kb < NSEQ; kb += 64) {
    // ---- stage K [key][d] : TDM 2D descriptor (64x64 halfs, pad 8/row) ----
#if HAVE_TDM
    if (wave == 0) {
      const unsigned long long ga =
          (unsigned long long)(const void*)(qkv + (size_t)(mbase + kb) * QKVLD + koff);
      // D# group0: count=1 | lds_addr | global_addr | type=2
      u32x4 g0 = { 1u, lds_k, (unsigned)ga, (unsigned)(ga >> 32) | 0x80000000u };
      // D# group1: data_size=2B, pad_enable, pad_interval=32dw, pad_amount=4dw,
      //            tensor_dim0=64, tensor_dim1=64, tile=64x64, dim0_stride=2304
      i32x8 g1 = { (int)((1u << 16) | (1u << 20) | (4u << 22) | (3u << 25)),
                   (int)(64u << 16), (int)(64u << 16), (int)(64u << 16),
                   64, QKVLD, 0, 0 };
      tdm_load_2d(g0, g1);
      __builtin_amdgcn_s_wait_tensorcnt(0);
    }
#else
    {
      const _Float16* kp = qkv + (size_t)(mbase + kb + srow) * QKVLD + koff + scb;
      *(uint4*)&sK[srow * LS + scb]     = *(const uint4*)(kp);
      *(uint4*)&sK[srow * LS + scb + 8] = *(const uint4*)(kp + 8);
    }
#endif
    // ---- stage V transposed [d][key] ----
    {
      const _Float16* vp = qkv + (size_t)(mbase + kb + srow) * QKVLD + voff + scb;
      U16h vv;
      vv.q[0] = *(const uint4*)(vp);
      vv.q[1] = *(const uint4*)(vp + 8);
#pragma unroll
      for (int j = 0; j < 16; ++j) sVT[(scb + j) * LS + srow] = vv.h[j];
    }
    __syncthreads();

    // ---- S = Q * K^T  (4 frags of 16x16, K-dim 64 in 2 steps) ----
    v8f s[4];
#pragma unroll
    for (int j = 0; j < 4; ++j) {
      v8f cacc = vzero;
#pragma unroll
      for (int kk = 0; kk < 2; ++kk) {
        FragA bk;
        const _Float16* p = &sK[(j * 16 + n16) * LS + kk * 32 + g * 8];
        bk.q[0] = *(const uint4*)(p);
        bk.q[1] = *(const uint4*)(p + 16);
        cacc = wmma32(qf[kk].v, bk.v, cacc);
      }
      s[j] = cacc;
    }

    // ---- online softmax (rows = rr + 8*g); reductions in pure VALU/DPP ----
    float mnew[8], alpha[8];
#pragma unroll
    for (int rr = 0; rr < 8; ++rr) {
      float mx = fmaxf(fmaxf(s[0][rr], s[1][rr]), fmaxf(s[2][rr], s[3][rr]));
      mx = red_max16(mx);
      mnew[rr]  = fmaxf(mrow[rr], mx);
      alpha[rr] = __expf(mrow[rr] - mnew[rr]);
      mrow[rr]  = mnew[rr];
    }
#pragma unroll
    for (int j = 0; j < 4; ++j)
#pragma unroll
      for (int rr = 0; rr < 8; ++rr)
        s[j][rr] = __expf(s[j][rr] - mnew[rr]);
#pragma unroll
    for (int rr = 0; rr < 8; ++rr) {
      float sm = (s[0][rr] + s[1][rr]) + (s[2][rr] + s[3][rr]);
      sm = red_sum16(sm);
      lrow[rr] = lrow[rr] * alpha[rr] + sm;
    }
#pragma unroll
    for (int c = 0; c < 4; ++c)
#pragma unroll
      for (int rr = 0; rr < 8; ++rr) o[c][rr] *= alpha[rr];

    // ---- bounce P through LDS: C-layout -> A-layout ----
#pragma unroll
    for (int j = 0; j < 4; ++j)
#pragma unroll
      for (int rr = 0; rr < 8; ++rr)
        myP[(rr + 8 * g) * LS + j * 16 + n16] = (_Float16)s[j][rr];

    // ---- O += P * V ----
#pragma unroll
    for (int kk = 0; kk < 2; ++kk) {
      FragA pa;
      const _Float16* pp = &myP[n16 * LS + kk * 32 + g * 8];
      pa.q[0] = *(const uint4*)(pp);
      pa.q[1] = *(const uint4*)(pp + 16);
#pragma unroll
      for (int c = 0; c < 4; ++c) {
        FragA vb;
        const _Float16* vp2 = &sVT[(c * 16 + n16) * LS + kk * 32 + g * 8];
        vb.q[0] = *(const uint4*)(vp2);
        vb.q[1] = *(const uint4*)(vp2 + 16);
        o[c] = wmma32(pa.v, vb.v, o[c]);
      }
    }
    __syncthreads();
  }

  // ---- normalize and store f16 attention output [8192][768] ----
  float inv[8];
#pragma unroll
  for (int rr = 0; rr < 8; ++rr) inv[rr] = 1.0f / lrow[rr];
#pragma unroll
  for (int c = 0; c < 4; ++c) {
    const int col = h * DHEAD + c * 16 + n16;
#pragma unroll
    for (int rr = 0; rr < 8; ++rr) {
      const int row = grow + rr + 8 * g;
      attn[(size_t)row * CDIM + col] = (_Float16)(o[c][rr] * inv[rr]);
    }
  }
}

// ---------------------------------------------------------------------------
extern "C" void kernel_launch(void* const* d_in, const int* in_sizes, int n_in,
                              void* d_out, int out_size, void* d_ws, size_t ws_size,
                              hipStream_t stream) {
  const float* x      = (const float*)d_in[0];
  const float* qkv_w  = (const float*)d_in[1];
  const float* qkv_b  = (const float*)d_in[2];
  const float* proj_w = (const float*)d_in[3];
  const float* proj_b = (const float*)d_in[4];
  float* out = (float*)d_out;

  _Float16* qkv_ws  = (_Float16*)d_ws;                       // [8192][2304] f16
  _Float16* attn_ws = qkv_ws + (size_t)MROWS * QKVLD;        // [8192][768]  f16

  // 1) QKV projection: x @ qkv_w^T + qkv_b -> f16 qkv ws
  {
    dim3 grid(QKVLD / 128, MROWS / 128);   // (18, 64)
    gemm_wmma<false, false><<<grid, dim3(256), 0, stream>>>(
        (const void*)x, qkv_w, qkv_b, (void*)qkv_ws, MROWS, CDIM, QKVLD);
  }
  // 2) Flash attention per (b,h) -> f16 attn ws
  {
    dim3 grid(BDIM * HHEADS, NSEQ / 128);  // (96, 8)
    flash_attn<<<grid, dim3(256), 0, stream>>>(qkv_ws, attn_ws);
  }
  // 3) Output projection: attn @ proj_w^T + proj_b -> fp32 out
  {
    dim3 grid(CDIM / 128, MROWS / 128);    // (6, 64)
    gemm_wmma<true, true><<<grid, dim3(256), 0, stream>>>(
        (const void*)attn_ws, proj_w, proj_b, (void*)out, MROWS, CDIM, CDIM);
  }
}
